// ARModel_57243324121405
// MI455X (gfx1250) — matrix operand order repaired and checked by reading, not compiled, not run on hardware
//
#include <hip/hip_runtime.h>
#include <hip/hip_bf16.h>
#include <math.h>

// ---------------------------------------------------------------------------
// Shapes: T=128, B=64, C=4, L=2048, F=64, H=45 conv channels.
// Conv: k=3, stride=2, pad=1, 5 layers, ReLU each, then mean over length.
// ---------------------------------------------------------------------------

typedef __attribute__((ext_vector_type(16))) __bf16 bf16x16;
typedef __attribute__((ext_vector_type(8)))  __bf16 bf16x8;
typedef __attribute__((ext_vector_type(8)))  float  floatx8;

#define NSAMP   8192      // T*B
#define CONV_H  45

// LDS layout (bytes):
//   wlds : [48][Kpad<=160] bf16 weights, kr-major K order       = 15360
//   blds : [48] f32 bias (zero padded)                          =   192
//   bufA : max(input [2050][16], y1 [514][48], y3 [130][48])    = 65600
//   bufB : max(y0 [1026][48], y2 [258][48], y4 [66][48])        = 98496
#define WLDS_OFF   0
#define BLDS_OFF   15360
#define BUFA_OFF   15552
#define BUFB_OFF   81152
#define SMEM_BYTES 179648

#define BUFA_ELEMS 32800   // 2050*16
#define BUFB_ELEMS 49248   // 1026*48

// ---------------------------------------------------------------------------
// Kernel 1: fused 5-layer conv encoder, one sample per workgroup.
// Layer l: GEMM A[48 x Kpad](bf16 weights) x B[Kpad x Lout](im2col view of
// channel-minor activations), f32 accum via v_wmma_f32_16x16x32_bf16.
//
// Activation LDS layout: act[row][Cpad] with row = l+1 (row 0 = conv zero-pad
// halo), channel-minor. Contraction order k = kr*Cpad + cin, so a fragment's
// two 8-value K runs are each 8 contiguous bf16: elem_off = (2*Cpad)*col + k0.
// ---------------------------------------------------------------------------
__global__ __launch_bounds__(256) void conv_encoder_kernel(
    const float* __restrict__ signals,
    const float* __restrict__ w0, const float* __restrict__ b0,
    const float* __restrict__ w1, const float* __restrict__ b1,
    const float* __restrict__ w2, const float* __restrict__ b2,
    const float* __restrict__ w3, const float* __restrict__ b3,
    const float* __restrict__ w4, const float* __restrict__ b4,
    float* __restrict__ conved)
{
  extern __shared__ __align__(16) char smem_raw[];
  __bf16* wlds = (__bf16*)(smem_raw + WLDS_OFF);
  float*  blds = (float*)(smem_raw + BLDS_OFF);
  __bf16* bufA = (__bf16*)(smem_raw + BUFA_OFF);
  __bf16* bufB = (__bf16*)(smem_raw + BUFB_OFF);

  const int n    = blockIdx.x;          // sample in [0, 8192)
  const int tid  = threadIdx.x;
  const int lane = tid & 31;            // wave32
  const int wave = tid >> 5;            // 8 waves / block
  const int half = lane >> 4;           // 0: lanes 0-15, 1: lanes 16-31
  const int ln16 = lane & 15;

  // Zero the whole input region [2050][16] (padding channels + halo rows must
  // be finite: WMMA propagates NaN even through zero weights).
  {
    uint32_t* p = (uint32_t*)bufA;
    for (int i = tid; i < BUFA_ELEMS / 2; i += 256) p[i] = 0u;
  }
  __syncthreads();

  // Stage input [4][2048] f32 -> bf16, channel-minor with halo: [(l+1)][16].
  // Global reads are fully coalesced (consecutive i = consecutive l).
  {
    const float* xin = signals + (size_t)n * (4 * 2048);
    for (int i = tid; i < 4 * 2048; i += 256) {
      const int cin = i >> 11;
      const int l   = i & 2047;
      bufA[(l + 1) * 16 + cin] = (__bf16)xin[i];
    }
  }

  const float* Ws[5]   = {w0, w1, w2, w3, w4};
  const float* Bs[5]   = {b0, b1, b2, b3, b4};
  const int    Cins[5] = {4, 45, 45, 45, 45};
  const int    Lins[5] = {2048, 1024, 512, 256, 128};

  __bf16* cur = bufA;
  __bf16* nxt = bufB;

#pragma unroll
  for (int layer = 0; layer < 5; ++layer) {
    const int Cin    = Cins[layer];
    const int Lin    = Lins[layer];
    const int Lout   = Lin >> 1;
    const int CpadK  = (layer == 0) ? 16 : 48;   // kr-block width in K order
    const int Kpad   = (layer == 0) ? 64 : 160;  // multiple of 32
    const int InStr  = (layer == 0) ? 16 : 48;   // cur row stride (channels)
    const float* __restrict__ W  = Ws[layer];
    const float* __restrict__ Bv = Bs[layer];

    // Pack weights: wlds[m][k], k = kr*CpadK + cin (zero padded -> no NaN,
    // and rows 45..47 / padded K produce exact zeros downstream).
    for (int i = tid; i < 48 * Kpad; i += 256) {
      const int m   = i / Kpad;
      const int k   = i - m * Kpad;
      const int kr  = k / CpadK;
      const int cin = k - kr * CpadK;
      wlds[i] = (m < CONV_H && kr < 3 && cin < Cin)
                  ? (__bf16)W[(m * Cin + cin) * 3 + kr] : (__bf16)0.0f;
    }
    if (tid < 48) blds[tid] = (tid < CONV_H) ? Bv[tid] : 0.0f;
    __syncthreads();   // weights + activations visible

    const int Ntn    = Lout >> 4;   // N tiles (power of two)
    const int ntiles = 3 * Ntn;     // 3 M-tiles (48 rows)

    for (int tile = wave; tile < ntiles; tile += 8) {
      const int mt  = tile / Ntn;
      const int nt  = tile - mt * Ntn;
      const int col = nt * 16 + ln16;

      // A: lanes 0-15/16-31 (half) hold K runs [kk+8h, +7] and [kk+16+8h, +7].
      const int base_a = (mt * 16 + ln16) * Kpad + half * 8;
      // B: lane holds column `col`, K runs [kk+8h, +7] and [kk+16+8h, +7]; with k = kr*CpadK+cin
      // and row = 2*col+kr, the element offset collapses to 2*InStr*col + k0.
      const int base_b = (2 * InStr) * col + half * 8;

      floatx8 acc = {};
#pragma unroll
      for (int kk = 0; kk < Kpad; kk += 32) {
        const bf16x8 a_lo = *(const bf16x8*)(wlds + base_a + kk);
        const bf16x8 a_hi = *(const bf16x8*)(wlds + base_a + kk + 16);
        const bf16x8 b_lo = *(const bf16x8*)(cur + base_b + kk);
        const bf16x8 b_hi = *(const bf16x8*)(cur + base_b + kk + 16);
        const bf16x16 afrag = __builtin_shufflevector(
            a_lo, a_hi, 0,1,2,3,4,5,6,7,8,9,10,11,12,13,14,15);
        const bf16x16 bfrag = __builtin_shufflevector(
            b_lo, b_hi, 0,1,2,3,4,5,6,7,8,9,10,11,12,13,14,15);
        acc = __builtin_amdgcn_wmma_f32_16x16x32_bf16(
            /*neg_a=*/false, afrag, /*neg_b=*/false, bfrag,
            /*c_mod=*/(short)0, acc, /*reuse_a=*/false, /*reuse_b=*/false);
      }

      // Epilogue: elem r -> M = mt*16 + 8*half + r. Bias from LDS (vector),
      // ReLU, pack 8 bf16, one 16-byte store (padding rows 45..47 are written
      // too -> exact zeros, keeping the next layer's B reads NaN-free).
      const floatx8 bias = *(const floatx8*)(blds + mt * 16 + half * 8);
      bf16x8 outv;
#pragma unroll
      for (int r = 0; r < 8; ++r) {
        float v = acc[r] + bias[r];
        v = v > 0.0f ? v : 0.0f;
        outv[r] = (__bf16)v;
      }
      *(bf16x8*)(nxt + (col + 1) * 48 + mt * 16 + half * 8) = outv;
    }

    // Zero output halo rows 0 and Lout+1 (disjoint from compute writes).
    if (tid < 24) {
      ((uint32_t*)nxt)[tid] = 0u;
    } else if (tid < 48) {
      ((uint32_t*)(nxt + (Lout + 1) * 48))[tid - 24] = 0u;
    }
    __syncthreads();   // layer done before wlds/nxt reuse

    __bf16* t = cur; cur = nxt; nxt = t;
  }

  // cur = y4, layout [66][48]; mean over 64 positions -> conved[n][45] f32.
  if (tid < CONV_H) {
    float s = 0.0f;
    for (int l = 0; l < 64; ++l) s += (float)cur[(l + 1) * 48 + tid];
    conved[(size_t)n * CONV_H + tid] = s * (1.0f / 64.0f);
  }
}

// ---------------------------------------------------------------------------
// Kernel 2: per-timestep BN(feats) + Linear(64->5), concat with conved,
// Linear(50->10) + BN + SiLU. One block per timestep; batch stats over B=64.
// ---------------------------------------------------------------------------
__global__ __launch_bounds__(256) void feat_combine_kernel(
    const float* __restrict__ feats, const float* __restrict__ conved,
    const float* __restrict__ bn1_g, const float* __restrict__ bn1_b,
    const float* __restrict__ fe_w,  const float* __restrict__ fe_b,
    const float* __restrict__ comb_w, const float* __restrict__ comb_b,
    const float* __restrict__ bn2_g, const float* __restrict__ bn2_b,
    float* __restrict__ z_out)
{
  const int t   = blockIdx.x;
  const int tid = threadIdx.x;

  __shared__ float fnorm[64 * 64];  // [b][f] normalized feats
  __shared__ float fe[64 * 5];      // [b][o]
  __shared__ float zpre[64 * 10];   // [b][j]
  __shared__ float mean2[10], rstd2[10];

  if (tid < 64) {
    const int f = tid;
    float m = 0.0f, s = 0.0f;
    for (int b = 0; b < 64; ++b) {
      const float v = feats[((size_t)t * 64 + b) * 64 + f];
      m += v; s += v * v;
    }
    m *= (1.0f / 64.0f);
    const float var = s * (1.0f / 64.0f) - m * m;   // biased variance
    const float rs  = rsqrtf(var + 1e-5f);
    const float g = bn1_g[f], bb = bn1_b[f];
    for (int b = 0; b < 64; ++b)
      fnorm[b * 64 + f] = (feats[((size_t)t * 64 + b) * 64 + f] - m) * rs * g + bb;
  }
  __syncthreads();

  for (int idx = tid; idx < 64 * 5; idx += blockDim.x) {
    const int b = idx / 5, o = idx - b * 5;
    float acc = fe_b[o];
    for (int f = 0; f < 64; ++f) acc += fnorm[b * 64 + f] * fe_w[o * 64 + f];
    fe[idx] = acc;
  }
  __syncthreads();

  for (int idx = tid; idx < 64 * 10; idx += blockDim.x) {
    const int b = idx / 10, j = idx - b * 10;
    float acc = comb_b[j];
    const float* w  = comb_w + j * 50;
    const float* cv = conved + ((size_t)t * 64 + b) * CONV_H;
    for (int i = 0; i < 45; ++i) acc += cv[i] * w[i];
    for (int i = 0; i < 5; ++i)  acc += fe[b * 5 + i] * w[45 + i];
    zpre[idx] = acc;
  }
  __syncthreads();

  if (tid < 10) {
    const int j = tid;
    float m = 0.0f, s = 0.0f;
    for (int b = 0; b < 64; ++b) {
      const float v = zpre[b * 10 + j];
      m += v; s += v * v;
    }
    m *= (1.0f / 64.0f);
    mean2[j] = m;
    rstd2[j] = rsqrtf(s * (1.0f / 64.0f) - m * m + 1e-5f);
  }
  __syncthreads();

  for (int idx = tid; idx < 64 * 10; idx += blockDim.x) {
    const int b = idx / 10, j = idx - b * 10;
    const float v = (zpre[idx] - mean2[j]) * rstd2[j] * bn2_g[j] + bn2_b[j];
    z_out[((size_t)t * 64 + b) * 10 + j] = v / (1.0f + __expf(-v));  // SiLU
  }
}

// ---------------------------------------------------------------------------
// Kernel 3: LSTM(10, hidden=3, proj=1) over T=128. Sequential, tiny:
// one thread per batch element, everything in registers.
// ---------------------------------------------------------------------------
__global__ __launch_bounds__(64) void lstm_kernel(
    const float* __restrict__ z,
    const float* __restrict__ w_ih, const float* __restrict__ w_hh,
    const float* __restrict__ b_ih, const float* __restrict__ b_hh,
    const float* __restrict__ w_hr, float* __restrict__ out)
{
  const int b = threadIdx.x;
  if (b >= 64) return;

  float wih[12][10], whh[12], bg[12], whr[3];
#pragma unroll
  for (int gi = 0; gi < 12; ++gi) {
    for (int i = 0; i < 10; ++i) wih[gi][i] = w_ih[gi * 10 + i];
    whh[gi] = w_hh[gi];
    bg[gi]  = b_ih[gi] + b_hh[gi];
  }
  whr[0] = w_hr[0]; whr[1] = w_hr[1]; whr[2] = w_hr[2];

  float h = 0.0f;
  float c[3] = {0.0f, 0.0f, 0.0f};

  for (int t = 0; t < 128; ++t) {
    const float* x = z + ((size_t)t * 64 + b) * 10;
    float g[12];
#pragma unroll
    for (int gi = 0; gi < 12; ++gi) {
      float a = bg[gi] + h * whh[gi];
      for (int i = 0; i < 10; ++i) a += x[i] * wih[gi][i];
      g[gi] = a;
    }
    float hv[3];
#pragma unroll
    for (int j = 0; j < 3; ++j) {
      const float ig = 1.0f / (1.0f + __expf(-g[j]));       // input
      const float fg = 1.0f / (1.0f + __expf(-g[3 + j]));   // forget
      const float gg = tanhf(g[6 + j]);                     // candidate
      const float og = 1.0f / (1.0f + __expf(-g[9 + j]));   // output
      c[j] = fg * c[j] + ig * gg;
      hv[j] = og * tanhf(c[j]);
    }
    h = hv[0] * whr[0] + hv[1] * whr[1] + hv[2] * whr[2];   // proj to 1
    out[t * 64 + b] = h;
  }
}

// ---------------------------------------------------------------------------
extern "C" void kernel_launch(void* const* d_in, const int* in_sizes, int n_in,
                              void* d_out, int out_size, void* d_ws, size_t ws_size,
                              hipStream_t stream) {
  (void)in_sizes; (void)n_in; (void)out_size; (void)ws_size;

  const float* signals = (const float*)d_in[0];
  const float* feats   = (const float*)d_in[1];
  const float* w0 = (const float*)d_in[2];  const float* b0 = (const float*)d_in[3];
  const float* w1 = (const float*)d_in[4];  const float* b1 = (const float*)d_in[5];
  const float* w2 = (const float*)d_in[6];  const float* b2 = (const float*)d_in[7];
  const float* w3 = (const float*)d_in[8];  const float* b3 = (const float*)d_in[9];
  const float* w4 = (const float*)d_in[10]; const float* b4 = (const float*)d_in[11];
  const float* bn1_g = (const float*)d_in[12]; const float* bn1_b = (const float*)d_in[13];
  const float* fe_w  = (const float*)d_in[14]; const float* fe_b  = (const float*)d_in[15];
  const float* comb_w = (const float*)d_in[16]; const float* comb_b = (const float*)d_in[17];
  const float* bn2_g  = (const float*)d_in[18]; const float* bn2_b  = (const float*)d_in[19];
  const float* w_ih = (const float*)d_in[20]; const float* w_hh = (const float*)d_in[21];
  const float* b_ih = (const float*)d_in[22]; const float* b_hh = (const float*)d_in[23];
  const float* w_hr = (const float*)d_in[24];

  // Workspace: conved [8192][45] f32, then z [128][64][10] f32 (~1.8 MB).
  float* conved = (float*)d_ws;
  float* z      = conved + (size_t)NSAMP * CONV_H;

  (void)hipFuncSetAttribute(reinterpret_cast<const void*>(conv_encoder_kernel),
                            hipFuncAttributeMaxDynamicSharedMemorySize, SMEM_BYTES);

  conv_encoder_kernel<<<NSAMP, 256, SMEM_BYTES, stream>>>(
      signals, w0, b0, w1, b1, w2, b2, w3, b3, w4, b4, conved);

  feat_combine_kernel<<<128, 256, 0, stream>>>(
      feats, conved, bn1_g, bn1_b, fe_w, fe_b, comb_w, comb_b, bn2_g, bn2_b, z);

  lstm_kernel<<<1, 64, 0, stream>>>(z, w_ih, w_hh, b_ih, b_hh, w_hr, (float*)d_out);
}
